// FusedSwitchExpert_69965017252291
// MI455X (gfx1250) — matrix operand order repaired and checked by reading, not compiled, not run on hardware
//
#include <hip/hip_runtime.h>

// ---------------------------------------------------------------------------
// Fused Switch-MoE expert FFN for gfx1250 (MI455X):
//   out = relu(x @ wi[e]^T) @ wo[e]^T   per expert token slice.
// bf16 operands (converted on the fly), f32 accumulation via
// V_WMMA_F32_16X16X32_BF16. One workgroup = 64 tokens x full d_model.
// 1024 threads = 32 wave32: 64 accum VGPRs/wave -> no scratch spills.
// ---------------------------------------------------------------------------

typedef __attribute__((ext_vector_type(16))) __bf16 v16bf;
typedef __attribute__((ext_vector_type(8)))  __bf16 bf16x8;
typedef __attribute__((ext_vector_type(4)))  __bf16 bf16x4;
typedef __attribute__((ext_vector_type(2)))  __bf16 bf16x2;
typedef __attribute__((ext_vector_type(8)))  float  v8f;
typedef __attribute__((ext_vector_type(4)))  float  f32x4;
typedef __attribute__((ext_vector_type(2)))  float  f32x2;

#define DMODEL 1024
#define DFF    4096
#define TM     64     // tokens per workgroup
#define KF     32     // d_ff slice per iteration
#define PX     1032   // padded bf16 row pitch for x / wi tiles (bank spread)
#define PW     40     // padded bf16 row pitch for wo tile and h tile

// LDS layout (bytes)
#define OFF_X   0                         // [64][PX]    bf16 = 132096
#define OFF_WI  132096                    // [32][PX]    bf16 = 66048
#define OFF_WO  198144                    // [1024][PW]  bf16 = 81920
#define OFF_HP  280064                    // [4][64][32] f32  = 32768
#define OFF_HB  312832                    // [64][PW]    bf16 = 5120
#define SMEM_BYTES 317952                 // 310.5 KB <= 320 KB per WGP

__device__ __forceinline__ bf16x4 cvt4(f32x4 v) {
  bf16x4 r;
  r.x = (__bf16)v.x; r.y = (__bf16)v.y; r.z = (__bf16)v.z; r.w = (__bf16)v.w;
  return r;
}

__device__ __forceinline__ v16bf cat16(bf16x8 lo, bf16x8 hi) {
  return __builtin_shufflevector(lo, hi, 0,1,2,3,4,5,6,7,8,9,10,11,12,13,14,15);
}

__global__ __launch_bounds__(1024, 1)
void moe_ffn_kernel(const float* __restrict__ X,
                    const float* __restrict__ WI,
                    const float* __restrict__ WO,
                    float* __restrict__ OUT) {
  extern __shared__ char smem[];
  __bf16* xs  = (__bf16*)(smem + OFF_X);   // x tile, bf16, pitch PX
  __bf16* wis = (__bf16*)(smem + OFF_WI);  // wi slice, bf16, pitch PX
  __bf16* wos = (__bf16*)(smem + OFF_WO);  // wo slice transposed, pitch PW
  float*  hp  = (float*) (smem + OFF_HP);  // split-K partials [4][64][32]
  __bf16* hb  = (__bf16*)(smem + OFF_HB);  // relu(h) bf16, pitch PW

  const int tid  = threadIdx.x;
  const int wave = tid >> 5;
  const int lane = tid & 31;
  const int lm   = lane & 15;   // lane within half (M or N index)
  const int lh   = lane >> 4;   // lane half (selects K / M sub-block)

  // ---- expert / token-tile lookup (capacities are compile-time constants)
  const int tiles[8] = {48, 16, 40, 24, 32, 32, 40, 24};          // C_e / 64
  const int offs[8]  = {0, 3072, 4096, 6656, 8192, 10240, 12288, 14848};
  int b = blockIdx.x, e = 0, acc0 = 0;
#pragma unroll
  for (int i = 0; i < 8; ++i) {
    if (b >= acc0 + tiles[i]) { acc0 += tiles[i]; e = i + 1; }
  }
  const int row0 = offs[e] + (b - acc0) * TM;
  const float* wiE = WI + (size_t)e * DFF * DMODEL;
  const float* woE = WO + (size_t)e * DMODEL * DFF;

  // ---- stage x tile (64 x 1024 f32 -> bf16 in LDS), coalesced b128 loads
  {
    const int r  = tid >> 4;          // 0..63
    const int cb = (tid & 15) * 64;   // 16 threads cover 1024 cols
    const float* src = X + (size_t)(row0 + r) * DMODEL + cb;
    __bf16* dst = xs + r * PX + cb;
#pragma unroll
    for (int i = 0; i < 16; ++i) {
      f32x4 v = *(const f32x4*)(src + i * 4);
      *(bf16x4*)(dst + i * 4) = cvt4(v);
    }
  }

  // ---- persistent output accumulators: 8 tiles of 16x16 f32 (64 VGPRs)
  v8f accd[8];
#pragma unroll
  for (int t = 0; t < 8; ++t) {
#pragma unroll
    for (int k = 0; k < 8; ++k) accd[t][k] = 0.0f;
  }

  // wave roles
  const int tA  = wave & 7;            // GEMM1 tile 0..7 (4 M x 2 F)
  const int kq  = wave >> 3;           // split-K quarter 0..3
  const int tmA = (tA & 3) * 16;       // GEMM1 M base
  const int tfA = (tA >> 2) * 16;      // GEMM1 F base
  const int m0B = (wave & 3) * 16;     // GEMM2 M strip
  const int n0B = (wave >> 2) * 128;   // GEMM2 N strip (8 tiles), 8 waves in N

  for (int f0 = 0; f0 < DFF; f0 += KF) {
    __syncthreads();   // protects wis/wos/hb from previous iteration readers

    // ---- stage wi slice: 32 rows x 1024 f32 -> bf16
    {
      const int fl = tid >> 5;          // 0..31
      const int cb = (tid & 31) * 32;   // 32 threads cover 1024 cols
      const float* src = wiE + (size_t)(f0 + fl) * DMODEL + cb;
      __bf16* dst = wis + fl * PX + cb;
#pragma unroll
      for (int i = 0; i < 8; ++i) {
        f32x4 v = *(const f32x4*)(src + i * 4);
        *(bf16x4*)(dst + i * 4) = cvt4(v);
      }
    }
    // ---- stage wo slice: 1024 rows x 32 f32 -> bf16, layout [d][kf]
    {
#pragma unroll
      for (int j = 0; j < 8; ++j) {
        const int g = tid + j * 1024;  // 8192 float4 groups total
        const int d = g >> 3;
        const int i = g & 7;
        f32x4 v = *(const f32x4*)(woE + (size_t)d * DFF + f0 + i * 4);
        *(bf16x4*)(wos + d * PW + i * 4) = cvt4(v);
      }
    }
    __syncthreads();

    // ---- GEMM1: h(64x32) = x(64x1024) @ wi_slice^T, split-K over 4 quarters
    {
      v8f c;
#pragma unroll
      for (int k = 0; k < 8; ++k) c[k] = 0.0f;
      const __bf16* arow = xs  + (tmA + lm) * PX;
      const __bf16* brow = wis + (tfA + lm) * PX;
#pragma unroll
      for (int kk = 0; kk < 8; ++kk) {
        const int kb = kq * 256 + kk * 32;
        // A 16x32 bf16: lane-half selects K sub-run of 8, +16 for VGPRs 4-7
        bf16x8 alo = *(const bf16x8*)(arow + kb + 8 * lh);
        bf16x8 ahi = *(const bf16x8*)(arow + kb + 8 * lh + 16);
        v16bf a = cat16(alo, ahi);
        // B 32x16 bf16: lane-half selects K sub-run of 8, +16 for VGPRs 4-7
        bf16x8 blo = *(const bf16x8*)(brow + kb + 8 * lh);
        bf16x8 bhi = *(const bf16x8*)(brow + kb + 8 * lh + 16);
        v16bf bm = cat16(blo, bhi);
        c = __builtin_amdgcn_wmma_f32_16x16x32_bf16(false, a, false, bm,
                                                    (short)0, c, false, false);
      }
      // C/D layout: VGPR r -> M = r + 8*lh, N = lm
#pragma unroll
      for (int r = 0; r < 8; ++r)
        hp[kq * 2048 + (tmA + r + 8 * lh) * KF + tfA + lm] = c[r];
    }
    __syncthreads();

    // ---- combine 4 split-K partials + ReLU + convert to bf16
    {
      const int m = tid >> 4;          // 0..63
      const int c = (tid & 15) * 2;    // 16 threads cover 32 cols
      f32x2 p0 = *(const f32x2*)(hp + m * KF + c);
      f32x2 p1 = *(const f32x2*)(hp + 2048 + m * KF + c);
      f32x2 p2 = *(const f32x2*)(hp + 4096 + m * KF + c);
      f32x2 p3 = *(const f32x2*)(hp + 6144 + m * KF + c);
      bf16x2 s;
      s.x = (__bf16)fmaxf((p0.x + p1.x) + (p2.x + p3.x), 0.0f);
      s.y = (__bf16)fmaxf((p0.y + p1.y) + (p2.y + p3.y), 0.0f);
      *(bf16x2*)(hb + m * PW + c) = s;
    }
    __syncthreads();

    // ---- GEMM2: out(64x1024) += h(64x32) @ wo_slice^T
    {
      const __bf16* arow = hb + (m0B + lm) * PW;
      bf16x8 alo = *(const bf16x8*)(arow + 8 * lh);
      bf16x8 ahi = *(const bf16x8*)(arow + 8 * lh + 16);
      v16bf a = cat16(alo, ahi);
#pragma unroll
      for (int nt = 0; nt < 8; ++nt) {
        const __bf16* brow = wos + (n0B + nt * 16 + lm) * PW;
        bf16x8 blo = *(const bf16x8*)(brow + 8 * lh);
        bf16x8 bhi = *(const bf16x8*)(brow + 8 * lh + 16);
        v16bf bm = cat16(blo, bhi);
        accd[nt] = __builtin_amdgcn_wmma_f32_16x16x32_bf16(false, a, false, bm,
                                                           (short)0, accd[nt],
                                                           false, false);
      }
    }
  }

  // ---- write out (f32), coalesced 16-lane row segments
#pragma unroll
  for (int nt = 0; nt < 8; ++nt) {
#pragma unroll
    for (int r = 0; r < 8; ++r) {
      OUT[(size_t)(row0 + m0B + r + 8 * lh) * DMODEL + n0B + nt * 16 + lm] =
          accd[nt][r];
    }
  }
}

extern "C" void kernel_launch(void* const* d_in, const int* in_sizes, int n_in,
                              void* d_out, int out_size, void* d_ws, size_t ws_size,
                              hipStream_t stream) {
  (void)in_sizes; (void)n_in; (void)out_size; (void)d_ws; (void)ws_size;
  const float* x  = (const float*)d_in[0];
  const float* wi = (const float*)d_in[1];
  const float* wo = (const float*)d_in[2];
  float* out = (float*)d_out;

  hipFuncSetAttribute(reinterpret_cast<const void*>(moe_ffn_kernel),
                      hipFuncAttributeMaxDynamicSharedMemorySize, SMEM_BYTES);
  moe_ffn_kernel<<<256, 1024, SMEM_BYTES, stream>>>(x, wi, wo, out);
}
